// Network_45191645888699
// MI455X (gfx1250) — matrix operand order repaired and checked by reading, not compiled, not run on hardware
//
#include <hip/hip_runtime.h>
#include <math.h>

#define C_DIM   128
#define KNN_K   16
#define NLAYER  5
#define NW      4      // waves per block in the transformer kernel
#define MAXP    12

typedef __attribute__((ext_vector_type(16))) __bf16 v16bf;
typedef __attribute__((ext_vector_type(8)))  float  v8f;
typedef __attribute__((__vector_size__(16))) int    v4i;

#if defined(__HIP_DEVICE_COMPILE__) && __has_builtin(__builtin_amdgcn_global_load_async_to_lds_b128)
#define HAVE_ASYNC 1
#else
#define HAVE_ASYNC 0
#endif

union AFrag { v16bf v; unsigned int u32[8]; unsigned short u16[16]; };

struct PairTab {
  int np; int total;
  int ctx_len[MAXP]; int tgt_off[MAXP]; int tgt_len[MAXP];
  int cum[MAXP + 1];
};

struct Params {
  const float *bx, *W_pe1, *b_pe1, *W_pe2, *b_pe2, *W_in, *b_in, *Wq, *Wk, *Wv,
              *W1, *b1, *W2, *b2, *W3, *b3;
  int* knn;
  unsigned short* wbf;   // 16 matrices of bf16, transposed [n][k], slot 0 = W_pe2
  float* out;
  int N;
};

__device__ __forceinline__ unsigned short f2bf(float x){
  union { float f; unsigned u; } v; v.f = x;
  unsigned r = v.u + 0x7FFFu + ((v.u >> 16) & 1u);
  return (unsigned short)(r >> 16);
}
__device__ __forceinline__ float bf2f(unsigned short h){
  union { unsigned u; float f; } v; v.u = ((unsigned)h) << 16; return v.f;
}

// A-fragment: 16x32 bf16 slice of row-major A[16][128], columns c0..c0+31.
// ISA layout: lane = m + 16*hi ; element e -> K = c0 + 16*(e>>3) + 8*hi + (e&7)
__device__ __forceinline__ void loadA(AFrag& a, const unsigned short* A, int m, int c0, int hi){
#pragma unroll
  for (int p = 0; p < 8; ++p){
    int e = 2 * p, g = e >> 3;
    a.u32[p] = *(const unsigned int*)&A[m * C_DIM + c0 + 16 * g + 8 * hi + (e & 7)];
  }
}
// B-fragment: 32x16 slice, reduction rows c0..c0+31, from transposed store Bt[n][k].
// ISA layout: lane = n + 16*hi ; element e -> K = c0 + 16*(e>>3) + 8*hi + (e&7)
__device__ __forceinline__ void loadB(AFrag& b, const unsigned short* Bt, int n, int c0, int hi){
#pragma unroll
  for (int p = 0; p < 8; ++p)
    b.u32[p] = *(const unsigned int*)&Bt[n * C_DIM + c0 + 16 * (p >> 2) + 8 * hi + 2 * (p & 3)];
}

// Stage one 128x128 bf16 matrix from global (already transposed) into LDS.
__device__ __forceinline__ void stage_mat(const unsigned short* gw, unsigned short* lw,
                                          int tid, int nthreads){
#if HAVE_ASYNC
  for (int i = tid * 8; i < C_DIM * C_DIM; i += nthreads * 8){
    v4i* gp = (v4i*)(gw + i);   // strip const, generic
    v4i* lp = (v4i*)(lw + i);
    __builtin_amdgcn_global_load_async_to_lds_b128(
        (__attribute__((address_space(1))) v4i*)gp,
        (__attribute__((address_space(3))) v4i*)lp, 0, 0);
  }
#else
  for (int i = tid * 8; i < C_DIM * C_DIM; i += nthreads * 8)
    *(uint4*)(lw + i) = *(const uint4*)(gw + i);
#endif
}
__device__ __forceinline__ void stage_fence(){
#if HAVE_ASYNC
  __builtin_amdgcn_s_wait_asynccnt(0);
#endif
}

// D[16x128] = A[16x128] @ B[128x128] (+bias) (+addend), B supplied transposed Bt[n][k].
__device__ void wave_gemm(const unsigned short* A, const unsigned short* Bt,
                          const float* bias, const unsigned short* addend,
                          unsigned short* out, bool outT, int lane)
{
  const int m = lane & 15, hi = lane >> 4;
  for (int nt = 0; nt < 8; ++nt){
    v8f acc = {};
#pragma unroll
    for (int kk = 0; kk < 4; ++kk){
      AFrag a, b;
      loadA(a, A, m, kk * 32, hi);
      loadB(b, Bt, nt * 16 + m, kk * 32, hi);
#if defined(__HIP_DEVICE_COMPILE__)
      acc = __builtin_amdgcn_wmma_f32_16x16x32_bf16(false, a.v, false, b.v,
                                                    (short)0, acc, false, false);
#endif
    }
    const int col = nt * 16 + m;
    const float bb = bias ? bias[col] : 0.f;
#pragma unroll
    for (int i = 0; i < 8; ++i){
      int r = i + 8 * hi;
      float v = acc[i] + bb;
      if (addend) v += bf2f(addend[r * C_DIM + col]);
      if (outT) out[col * 16 + r]    = f2bf(v);
      else      out[r * C_DIM + col] = f2bf(v);
    }
  }
}

__global__ void zero_kernel(float* o){ if (threadIdx.x == 0) o[0] = 0.f; }

// ---- one-time weight prep: f32 -> bf16, transposed [n][k]; slot0=W_pe2, then q/k/v per layer
__global__ void prep_kernel(Params prm){
  int i = blockIdx.x * blockDim.x + threadIdx.x;
  const int total = 16 * C_DIM * C_DIM;
  if (i >= total) return;
  int mat = i >> 14;
  int r   = i & 16383;
  int c = r >> 7, n = r & 127;
  const float* src;
  if (mat == 0) src = prm.W_pe2;
  else {
    int l = (mat - 1) / 3, which = (mat - 1) % 3;
    src = (which == 0 ? prm.Wq : which == 1 ? prm.Wk : prm.Wv) + (size_t)l * C_DIM * C_DIM;
  }
  prm.wbf[(size_t)mat * 16384 + n * C_DIM + c] = f2bf(src[r]);
}

// ---------------- KNN: one wave per target, brute force + top-16 merge ----------------
__global__ void knn_kernel(Params prm, PairTab tab)
{
  const int lane = threadIdx.x & 31;
  const int w    = threadIdx.x >> 5;
  const int t    = blockIdx.x * 4 + w;
  if (t >= 2 * tab.total) return;
  const int b = t / tab.total, mt = t % tab.total;
  int p = 0;
  while (p + 1 < tab.np && mt >= tab.cum[p + 1]) ++p;
  const int tp = tab.tgt_off[p] + (mt - tab.cum[p]);
  const int cn = tab.ctx_len[p];
  const float* base = prm.bx + (size_t)b * prm.N * 6;
  const float tx = base[tp * 6], ty = base[tp * 6 + 1], tz = base[tp * 6 + 2];

  float bd[16]; int bi[16];
#pragma unroll
  for (int i = 0; i < 16; ++i){ bd[i] = 3.4e38f; bi[i] = 0; }

  for (int j = lane; j < cn; j += 32){
    float dx = base[j * 6] - tx, dy = base[j * 6 + 1] - ty, dz = base[j * 6 + 2] - tz;
    float d = dx * dx + dy * dy + dz * dz;
    if (d < bd[15]){
      bd[15] = d; bi[15] = j;
#pragma unroll
      for (int s2 = 15; s2 > 0; --s2){
        if (bd[s2] < bd[s2 - 1]){
          float td = bd[s2]; bd[s2] = bd[s2 - 1]; bd[s2 - 1] = td;
          int   ti = bi[s2]; bi[s2] = bi[s2 - 1]; bi[s2 - 1] = ti;
        }
      }
    }
  }

  int* outp = prm.knn + ((size_t)b * tab.total + mt) * KNN_K;
  for (int kk = 0; kk < KNN_K; ++kk){
    float d = bd[0]; int ii = bi[0]; int src = lane;
#pragma unroll
    for (int msk = 16; msk >= 1; msk >>= 1){
      float od = __shfl_xor(d, msk, 32);
      int   oi = __shfl_xor(ii, msk, 32);
      int   os = __shfl_xor(src, msk, 32);
      if (od < d || (od == d && os < src)){ d = od; ii = oi; src = os; }
    }
    if (lane == 0) outp[kk] = ii;
    if (lane == src){
#pragma unroll
      for (int s2 = 0; s2 < 15; ++s2){ bd[s2] = bd[s2 + 1]; bi[s2] = bi[s2 + 1]; }
      bd[15] = 3.4e38f;
    }
  }
}

// ---------------- Point transformer: one wave per target, WMMA bf16 ----------------
__global__ void __launch_bounds__(NW * 32, 1) pt_kernel(Params prm, PairTab tab)
{
  __shared__ __align__(16) unsigned short sWq[C_DIM * C_DIM];   // transposed bf16 weights
  __shared__ __align__(16) unsigned short sWk[C_DIM * C_DIM];
  __shared__ __align__(16) unsigned short sWv[C_DIM * C_DIM];
  __shared__ __align__(16) unsigned short sF [NW][16 * C_DIM];  // features (bf16, row-major)
  __shared__ __align__(16) unsigned short sPe[NW][16 * C_DIM];  // positional encoding
  __shared__ __align__(16) unsigned short sQ [NW][16 * C_DIM];
  __shared__ __align__(16) unsigned short sK [NW][16 * C_DIM];
  __shared__ __align__(16) unsigned short sV [NW][16 * C_DIM];  // (v+pe) transposed [c][j]
  __shared__ __align__(16) unsigned short sAt[NW][16 * 16];     // attention probs
  __shared__ float sGeo[NW][16 * 4];
  __shared__ float sAtt[NW][16 * 4];
  __shared__ float sHd [NW][224];                               // feat|h1|h2|ms

  const int tid  = threadIdx.x;
  const int lane = tid & 31;
  const int w    = tid >> 5;
  const int t    = blockIdx.x * NW + w;
  const bool active = (t < 2 * tab.total);

  int b = 0, tp = 0, mt = 0;
  const float* base = prm.bx;
  if (active){
    b = t / tab.total; mt = t % tab.total;
    int p = 0;
    while (p + 1 < tab.np && mt >= tab.cum[p + 1]) ++p;
    tp = tab.tgt_off[p] + (mt - tab.cum[p]);
    base = prm.bx + (size_t)b * prm.N * 6;
  }

  // -------- prologue: stage W_pe2^T (bf16, prepped), geo/attr, h1, f, pe --------
  stage_mat(prm.wbf, sWq, tid, blockDim.x);
  stage_fence();
  __syncthreads();

  if (active){
    const int* idx = prm.knn + ((size_t)b * tab.total + mt) * KNN_K;
    const float tx = base[tp * 6], ty = base[tp * 6 + 1], tz = base[tp * 6 + 2];
    float nrm = 0.f;
    if (lane < KNN_K){
      int ci = idx[lane];
      float gx = base[ci * 6]     - tx;
      float gy = base[ci * 6 + 1] - ty;
      float gz = base[ci * 6 + 2] - tz;
      sGeo[w][lane * 4 + 0] = gx; sGeo[w][lane * 4 + 1] = gy; sGeo[w][lane * 4 + 2] = gz;
      sAtt[w][lane * 4 + 0] = base[ci * 6 + 3] * (1.f / 255.f);
      sAtt[w][lane * 4 + 1] = base[ci * 6 + 4] * (1.f / 511.f);
      sAtt[w][lane * 4 + 2] = base[ci * 6 + 5] * (1.f / 511.f);
      nrm = sqrtf(gx * gx + gy * gy + gz * gz);
    }
#pragma unroll
    for (int msk = 8; msk >= 1; msk >>= 1) nrm = fmaxf(nrm, __shfl_xor(nrm, msk, 32));
    const float inv = (nrm > 0.f) ? 1.f / nrm : 0.f;
    if (lane < KNN_K){
      sGeo[w][lane * 4 + 0] *= inv; sGeo[w][lane * 4 + 1] *= inv; sGeo[w][lane * 4 + 2] *= inv;
    }
    // h1 = relu(geo @ W_pe1 + b_pe1) -> sQ ; f = attr @ W_in + b_in -> sF
    for (int q = 0; q < (16 * C_DIM) / 32; ++q){
      int o = q * 32 + lane;
      int mm = o >> 7, c = o & 127;
      float g0 = sGeo[w][mm * 4], g1 = sGeo[w][mm * 4 + 1], g2 = sGeo[w][mm * 4 + 2];
      float h = prm.b_pe1[c] + g0 * prm.W_pe1[c] + g1 * prm.W_pe1[C_DIM + c] + g2 * prm.W_pe1[2 * C_DIM + c];
      sQ[w][mm * C_DIM + c] = f2bf(fmaxf(h, 0.f));
      float a0 = sAtt[w][mm * 4], a1 = sAtt[w][mm * 4 + 1], a2 = sAtt[w][mm * 4 + 2];
      float f0 = prm.b_in[c] + a0 * prm.W_in[c] + a1 * prm.W_in[C_DIM + c] + a2 * prm.W_in[2 * C_DIM + c];
      sF[w][mm * C_DIM + c] = f2bf(f0);
    }
    wave_gemm(sQ[w], sWq, prm.b_pe2, nullptr, sPe[w], false, lane);  // pe
  }

  // -------- attention layers --------
  for (int l = 0; l < NLAYER; ++l){
    __syncthreads();
    const unsigned short* gw = prm.wbf + (size_t)(1 + 3 * l) * 16384;
    if (l + 1 < NLAYER)
      __builtin_prefetch(prm.wbf + (size_t)(1 + 3 * (l + 1)) * 16384, 0, 1);
    stage_mat(gw,             sWq, tid, blockDim.x);
    stage_mat(gw + 16384,     sWk, tid, blockDim.x);
    stage_mat(gw + 2 * 16384, sWv, tid, blockDim.x);
    stage_fence();
    __syncthreads();
    if (active){
      wave_gemm(sF[w], sWq, nullptr, nullptr, sQ[w], false, lane);
      wave_gemm(sF[w], sWk, nullptr, nullptr, sK[w], false, lane);
      wave_gemm(sF[w], sWv, nullptr, sPe[w],  sV[w], true,  lane);  // (v+pe)^T

      const int m = lane & 15, hi = lane >> 4;
      // logits = q @ k^T  (row-major k serves directly as transposed B operand)
      v8f s = {};
#pragma unroll
      for (int kk = 0; kk < 4; ++kk){
        AFrag a, bb;
        loadA(a, sQ[w], m, kk * 32, hi);
        loadB(bb, sK[w], m, kk * 32, hi);
#if defined(__HIP_DEVICE_COMPILE__)
        s = __builtin_amdgcn_wmma_f32_16x16x32_bf16(false, a.v, false, bb.v,
                                                    (short)0, s, false, false);
#endif
      }
      const float scl = 0.08838834764831845f;  // 1/sqrt(128)
#pragma unroll
      for (int i = 0; i < 8; ++i){
        float x = s[i] * scl;
        float mx = x;
#pragma unroll
        for (int msk = 8; msk >= 1; msk >>= 1) mx = fmaxf(mx, __shfl_xor(mx, msk, 32));
        float e = __expf(x - mx);
        float sm = e;
#pragma unroll
        for (int msk = 8; msk >= 1; msk >>= 1) sm += __shfl_xor(sm, msk, 32);
        sAt[w][(i + 8 * hi) * 16 + m] = f2bf(e / sm);
      }
      // f += attn @ (v+pe)  (K padded 16 -> 32 with zeros)
      for (int nt = 0; nt < 8; ++nt){
        const int col = nt * 16 + m;
        v8f acc;
#pragma unroll
        for (int i = 0; i < 8; ++i) acc[i] = bf2f(sF[w][(i + 8 * hi) * C_DIM + col]);
        AFrag a, bb;
#pragma unroll
        for (int p = 0; p < 4; ++p){
          a.u32[p]     = *(const unsigned int*)&sAt[w][m * 16 + 8 * hi + 2 * p];
          a.u32[p + 4] = 0u;
        }
#pragma unroll
        for (int p = 0; p < 8; ++p)
          bb.u32[p] = (p >= 4) ? 0u : *(const unsigned int*)&sV[w][col * 16 + 8 * hi + 2 * p];
#if defined(__HIP_DEVICE_COMPILE__)
        acc = __builtin_amdgcn_wmma_f32_16x16x32_bf16(false, a.v, false, bb.v,
                                                      (short)0, acc, false, false);
#endif
#pragma unroll
        for (int i = 0; i < 8; ++i) sF[w][(i + 8 * hi) * C_DIM + col] = f2bf(acc[i]);
      }
    }
  }

  // -------- head MLP + bits --------
  if (active){
#pragma unroll
    for (int tg = 0; tg < 4; ++tg){
      int c = lane * 4 + tg;
      float mx = -3.4e38f;
      for (int r = 0; r < 16; ++r) mx = fmaxf(mx, bf2f(sF[w][r * C_DIM + c]));
      sHd[w][c] = mx;
    }
#pragma unroll
    for (int hh = 0; hh < 2; ++hh){
      int o = lane + 32 * hh;
      float s0 = prm.b1[o];
      for (int c = 0; c < C_DIM; ++c) s0 += sHd[w][c] * prm.W1[c * 64 + o];
      sHd[w][128 + o] = fmaxf(s0, 0.f);
    }
    if (lane < 16){
      float s0 = prm.b2[lane];
      for (int j = 0; j < 64; ++j) s0 += sHd[w][128 + j] * prm.W2[j * 16 + lane];
      sHd[w][192 + lane] = fmaxf(s0, 0.f);
    }
    if (lane < 6){
      float s0 = prm.b3[lane];
      for (int j = 0; j < 16; ++j) s0 += sHd[w][192 + j] * prm.W3[j * 6 + lane];
      sHd[w][208 + lane] = s0;
    }
    float bits = 0.f;
    if (lane < 3){
      float mu = (sHd[w][208 + lane] + 0.5f) * 255.f;
      float sg = __expf(sHd[w][211 + lane]) * 32.f;
      sg = fminf(fmaxf(sg, 1e-10f), 1e10f);
      float x  = base[tp * 6 + 3 + lane];
      float iv = 1.f / (sg * 1.41421356237f);
      float ch = 0.5f * (1.f + erff((x + 0.5f - mu) * iv));
      float cl = 0.5f * (1.f + erff((x - 0.5f - mu) * iv));
      float pb = ch - cl + 1e-10f;
      bits = fminf(fmaxf(-log2f(pb), 0.f), 50.f);
    }
    bits += __shfl_xor(bits, 1, 32);
    bits += __shfl_xor(bits, 2, 32);
    if (lane == 0) atomicAdd(prm.out, bits);
  }
}

extern "C" void kernel_launch(void* const* d_in, const int* in_sizes, int n_in,
                              void* d_out, int out_size, void* d_ws, size_t ws_size,
                              hipStream_t stream)
{
  (void)n_in; (void)out_size; (void)ws_size;
  Params prm;
  prm.bx    = (const float*)d_in[0];
  prm.W_pe1 = (const float*)d_in[1];
  prm.b_pe1 = (const float*)d_in[2];
  prm.W_pe2 = (const float*)d_in[3];
  prm.b_pe2 = (const float*)d_in[4];
  prm.W_in  = (const float*)d_in[5];
  prm.b_in  = (const float*)d_in[6];
  prm.Wq    = (const float*)d_in[7];
  prm.Wk    = (const float*)d_in[8];
  prm.Wv    = (const float*)d_in[9];
  prm.W1    = (const float*)d_in[10];
  prm.b1    = (const float*)d_in[11];
  prm.W2    = (const float*)d_in[12];
  prm.b2    = (const float*)d_in[13];
  prm.W3    = (const float*)d_in[14];
  prm.b3    = (const float*)d_in[15];
  prm.out   = (float*)d_out;
  const int B = 2;
  prm.N = in_sizes[0] / (B * 6);

  // Pair schedule (reference constants: granularity=4096, init_ratio=128, expand_ratio=2)
  PairTab tab{};
  const int gran = 4096, initr = 128, ex = 2;
  int bsz = prm.N / initr; if (bsz > gran) bsz = gran;
  int wsz = bsz, cursor = bsz, tot = 0;
  tab.np = 0; tab.cum[0] = 0;
  while (cursor < prm.N && tab.np < MAXP){
    wsz = wsz * ex; if (wsz > gran) wsz = gran;
    int tl = prm.N - cursor; if (tl > wsz) tl = wsz;
    tab.ctx_len[tab.np] = cursor;
    tab.tgt_off[tab.np] = cursor;
    tab.tgt_len[tab.np] = tl;
    tot += tl;
    tab.np++;
    tab.cum[tab.np] = tot;
    cursor += wsz;
  }
  tab.total = tot;

  // workspace layout: [ knn indices | bf16 transposed weights (16 * 16384) ]
  size_t knn_bytes = (size_t)B * tab.total * KNN_K * sizeof(int);
  knn_bytes = (knn_bytes + 255) & ~(size_t)255;
  prm.knn = (int*)d_ws;
  prm.wbf = (unsigned short*)((char*)d_ws + knn_bytes);

  const int nt = 2 * tab.total;
  hipLaunchKernelGGL(zero_kernel, dim3(1), dim3(32), 0, stream, prm.out);
  hipLaunchKernelGGL(prep_kernel, dim3((16 * C_DIM * C_DIM + 255) / 256), dim3(256), 0, stream, prm);
  hipLaunchKernelGGL(knn_kernel, dim3((nt + 3) / 4), dim3(128), 0, stream, prm, tab);
  hipLaunchKernelGGL(pt_kernel, dim3((nt + NW - 1) / NW), dim3(NW * 32), 0, stream, prm, tab);
}
